// MaskedConvFlow_23184233464248
// MI455X (gfx1250) — matrix operand order repaired and checked by reading, not compiled, not run on hardware
//
#include <hip/hip_runtime.h>
#include <hip/hip_bf16.h>
#include <math.h>

// ---------------------------------------------------------------------------
// MaskedConvFlow fused kernel for gfx1250 (MI455X).
//   conv1 (2x3 shifted, 32->64) + ELU + conv2 (1x1, 64->64) + affine flow
// GEMM mapping (per wave, wave32):
//   conv1: A[16 pix x 192] * B[192 x 64] -> 6 K-steps x 4 N-tiles of
//          v_wmma_f32_16x16x32_f16
//   conv2: A[16 pix x 64]  * B[64 x 64]  -> 2 K-steps x 4 N-tiles
// A-matrix comes from an im2col LDS tile xe[col][k] so every A fragment is
// two contiguous ds_load_b128 per lane (no per-element index math in the
// hot loop).  B fragments are pre-packed in d_ws in per-lane WMMA layout.
// ---------------------------------------------------------------------------

typedef __attribute__((ext_vector_type(16))) _Float16 v16h;
typedef __attribute__((ext_vector_type(8)))  _Float16 v8h;
typedef __attribute__((ext_vector_type(8)))  float    v8f;

#define BB   8
#define CC   32
#define HH   256
#define WW   256
#define HID  64
#define K1   192          // C*KH*KW = 32*2*3
#define XSTR 200          // xe k-stride in halfs (400 B: 16B-aligned, bank-friendly)
#define HSTR 72           // Hs o-stride in halfs (144 B: 16B-aligned, bank-friendly)

// workspace layout (bytes)
#define OFF_W1   0                       // 12288 halfs = 24576 B
#define OFF_W2   24576                   //  4096 halfs =  8192 B
#define OFF_B1   32768                   //    64 floats
#define OFF_B2   33024                   //    64 floats
#define OFF_PART 33280                   //  4096 floats
#define OUT_ELEMS (BB*CC*HH*WW)          // 16777216

// ---------------------------------------------------------------------------
// Prep: weight-norm scales, f16 conversion, WMMA B-fragment packing.
// Fragment f = (kstep*4 + ntile); lane L: column n = L&15, kgroup = L>>4,
// element i (0..15): k = kstep*32 + 8*kgroup + (i&7) + 16*(i>>3).
// ---------------------------------------------------------------------------
__global__ void mcf_prep_kernel(const float* __restrict__ v1,
                                const float* __restrict__ g1,
                                const float* __restrict__ b1,
                                const float* __restrict__ v2,
                                const float* __restrict__ g2,
                                const float* __restrict__ b2,
                                _Float16* __restrict__ w1h,
                                _Float16* __restrict__ w2h,
                                float* __restrict__ b1f,
                                float* __restrict__ b2f) {
  __shared__ float s1[HID];
  __shared__ float s2[2*CC];
  int t = threadIdx.x;
  if (t < HID) {                       // conv1 weight-norm scale
    float s = 0.f;
    for (int k = 0; k < K1; ++k) { float v = v1[t*K1 + k]; s += v*v; }
    s1[t] = g1[t] * rsqrtf(s);
  } else if (t < 128) {                // conv2 weight-norm scale
    int o = t - 64;
    float s = 0.f;
    for (int k = 0; k < HID; ++k) { float v = v2[o*HID + k]; s += v*v; }
    s2[o] = g2[o] * rsqrtf(s);
  } else if (t < 192) {
    b1f[t - 128] = b1[t - 128];
  } else {
    b2f[t - 192] = b2[t - 192];
  }
  __syncthreads();
  // conv1 fragments: 6 ksteps * 4 ntiles * 32 lanes * 16 halfs = 12288
  for (int idx = t; idx < 6*4*32*16; idx += 256) {
    int i    = idx & 15;
    int lane = (idx >> 4) & 31;
    int nb   = (idx >> 9) & 3;
    int kk   = idx >> 11;
    int o = nb*16 + (lane & 15);
    int k = kk*32 + (lane >> 4)*8 + (i & 7) + (i >> 3)*16;          // < 192
    w1h[idx] = (_Float16)(v1[o*K1 + k] * s1[o]);
  }
  // conv2 fragments: 2 ksteps * 4 ntiles * 32 lanes * 16 halfs = 4096
  for (int idx = t; idx < 2*4*32*16; idx += 256) {
    int i    = idx & 15;
    int lane = (idx >> 4) & 31;
    int nb   = (idx >> 9) & 3;
    int kk   = idx >> 11;
    int o = nb*16 + (lane & 15);
    int k = kk*32 + (lane >> 4)*8 + (i & 7) + (i >> 3)*16;          // < 64
    w2h[idx] = (_Float16)(v2[o*HID + k] * s2[o]);
  }
}

// ---------------------------------------------------------------------------
// Main fused kernel: grid = B*H*2, block = 256 (8 waves, 16 pixels each).
// ---------------------------------------------------------------------------
__global__ void __launch_bounds__(256)
mcf_main_kernel(const float* __restrict__ x,
                const _Float16* __restrict__ w1h,
                const _Float16* __restrict__ w2h,
                const float* __restrict__ b1f,
                const float* __restrict__ b2f,
                float* __restrict__ out,
                float* __restrict__ part) {
  __shared__ _Float16 xe[128*XSTR];           // im2col tile [col][k], f16
  __shared__ _Float16 Hs[8*16*HSTR];          // per-wave hidden re-layout
  __shared__ float    red[256];

  const int tid  = threadIdx.x;
  const int lane = tid & 31;
  const int wid  = tid >> 5;
  const int bid  = blockIdx.x;
  const int b    = bid >> 9;
  const int h    = (bid >> 1) & 255;
  const int wbase = (bid & 1) * 128;          // 128-col half-row per block

  // ---- im2col staging: xe[col][k] = x[b][c][h-2+kh][wbase+col-1+kw] ----
  // k = c*6 + kh*3 + kw  (matches B-fragment packing).
  for (int idx = tid; idx < K1*128; idx += 256) {
    int col = idx & 127;
    int k   = idx >> 7;                       // 0..191
    int c   = k / 6;
    int r6  = k - c*6;
    int kh  = r6 / 3;
    int kw  = r6 - kh*3;
    int row = h - 2 + kh;                     // shifted conv: rows h-2,h-1
    int w   = wbase + col - 1 + kw;
    float v = 0.f;
    if (row >= 0 && (unsigned)w < WW)
      v = x[((b*CC + c)*HH + row)*WW + w];
    xe[col*XSTR + k] = (_Float16)v;
  }
  __syncthreads();

  const int m  = lane & 15;                   // pixel within M-tile
  const int kg = lane >> 4;                   // K-half select
  const int w0 = wbase + wid*16;

  // ---- conv1: 6 K-steps x 4 N-tiles of WMMA f16 -> f32 ----
  const _Float16* ap = xe + (wid*16 + m)*XSTR + 8*kg;
  v8f acc[4] = {};
  #pragma unroll
  for (int kk = 0; kk < 6; ++kk) {
    v8h lo = *(const v8h*)(ap + kk*32);       // k = kk*32 + 8*kg + 0..7
    v8h hi = *(const v8h*)(ap + kk*32 + 16);  // k = kk*32 + 16 + 8*kg + 0..7
    v16h a = __builtin_shufflevector(lo, hi, 0,1,2,3,4,5,6,7,
                                             8,9,10,11,12,13,14,15);
    #pragma unroll
    for (int nb = 0; nb < 4; ++nb) {
      v16h bf = *(const v16h*)(w1h + ((kk*4 + nb)*32 + lane)*16);
      acc[nb] = __builtin_amdgcn_wmma_f32_16x16x32_f16(
          false, a, false, bf, (short)0, acc[nb], false, false);
    }
  }

  // ---- bias + ELU, re-layout D(16x64) -> LDS for conv2 A-fragments ----
  {
    _Float16* hw = Hs + wid*16*HSTR;
    #pragma unroll
    for (int nb = 0; nb < 4; ++nb) {
      int o = nb*16 + m;
      float bv = b1f[o];
      #pragma unroll
      for (int r = 0; r < 8; ++r) {
        float hv = acc[nb][r] + bv;
        float e  = hv > 0.f ? hv : (__expf(hv) - 1.f);   // ELU
        hw[(8*kg + r)*HSTR + o] = (_Float16)e;
      }
    }
  }
  __syncthreads();

  // ---- conv2 (1x1, 64->64): 2 K-steps x 4 N-tiles ----
  v8f acc2[4] = {};
  {
    const _Float16* hr = Hs + (wid*16 + m)*HSTR + 8*kg;
    #pragma unroll
    for (int kk = 0; kk < 2; ++kk) {
      v8h lo = *(const v8h*)(hr + kk*32);
      v8h hi = *(const v8h*)(hr + kk*32 + 16);
      v16h a = __builtin_shufflevector(lo, hi, 0,1,2,3,4,5,6,7,
                                               8,9,10,11,12,13,14,15);
      #pragma unroll
      for (int nb = 0; nb < 4; ++nb) {
        v16h bf = *(const v16h*)(w2h + ((kk*4 + nb)*32 + lane)*16);
        acc2[nb] = __builtin_amdgcn_wmma_f32_16x16x32_f16(
            false, a, false, bf, (short)0, acc2[nb], false, false);
      }
    }
  }

  // ---- epilogue: mu (ch 0..31) + sigmoid(log_scale+2) (ch 32..63) ----
  float ldsum = 0.f;
  #pragma unroll
  for (int p = 0; p < 2; ++p) {
    int c = p*16 + m;
    float bmu = b2f[c];
    float bls = b2f[c + CC] + 2.0f;
    int base = ((b*CC + c)*HH + h)*WW + w0 + 8*kg;   // 8 consecutive w's
    float4 xa = *(const float4*)(x + base);
    float4 xb = *(const float4*)(x + base + 4);
    float xv[8] = {xa.x, xa.y, xa.z, xa.w, xb.x, xb.y, xb.z, xb.w};
    float o8[8];
    #pragma unroll
    for (int r = 0; r < 8; ++r) {
      float mu = acc2[p][r] + bmu;
      float sc = 1.f / (1.f + __expf(-(acc2[p + 2][r] + bls)));
      o8[r] = xv[r]*sc + mu;
      ldsum += __logf(sc);
    }
    *(float4*)(out + base)     = make_float4(o8[0], o8[1], o8[2], o8[3]);
    *(float4*)(out + base + 4) = make_float4(o8[4], o8[5], o8[6], o8[7]);
  }

  // ---- deterministic block reduction of log(scale) ----
  red[tid] = ldsum;
  __syncthreads();
  #pragma unroll
  for (int s = 128; s > 0; s >>= 1) {
    if (tid < s) red[tid] += red[tid + s];
    __syncthreads();
  }
  if (tid == 0) part[bid] = red[0];
}

// ---------------------------------------------------------------------------
// Final: deterministic reduce of 512 block-partials per batch into logdet[b].
// ---------------------------------------------------------------------------
__global__ void mcf_reduce_kernel(const float* __restrict__ part,
                                  float* __restrict__ logdet) {
  __shared__ float red[256];
  int b = blockIdx.x, t = threadIdx.x;
  red[t] = part[b*512 + t] + part[b*512 + 256 + t];
  __syncthreads();
  #pragma unroll
  for (int s = 128; s > 0; s >>= 1) {
    if (t < s) red[t] += red[t + s];
    __syncthreads();
  }
  if (t == 0) logdet[b] = red[0];
}

// ---------------------------------------------------------------------------
extern "C" void kernel_launch(void* const* d_in, const int* in_sizes, int n_in,
                              void* d_out, int out_size, void* d_ws, size_t ws_size,
                              hipStream_t stream) {
  const float* x  = (const float*)d_in[0];
  const float* v1 = (const float*)d_in[1];
  const float* g1 = (const float*)d_in[2];
  const float* b1 = (const float*)d_in[3];
  const float* v2 = (const float*)d_in[4];
  const float* g2 = (const float*)d_in[5];
  const float* b2 = (const float*)d_in[6];
  float* out = (float*)d_out;

  char* ws = (char*)d_ws;
  _Float16* w1h = (_Float16*)(ws + OFF_W1);
  _Float16* w2h = (_Float16*)(ws + OFF_W2);
  float*    b1f = (float*)(ws + OFF_B1);
  float*    b2f = (float*)(ws + OFF_B2);
  float*    prt = (float*)(ws + OFF_PART);

  mcf_prep_kernel<<<1, 256, 0, stream>>>(v1, g1, b1, v2, g2, b2,
                                         w1h, w2h, b1f, b2f);
  mcf_main_kernel<<<BB*HH*2, 256, 0, stream>>>(x, w1h, w2h, b1f, b2f,
                                               out, prt);
  mcf_reduce_kernel<<<BB, 256, 0, stream>>>(prt, out + OUT_ELEMS);
}
